// GMTRouterModel_86139864089178
// MI455X (gfx1250) — compile-verified, matrix operand order checked by reading, not run on hardware
//
#include <hip/hip_runtime.h>
#include <hip/hip_bf16.h>

// ---- problem constants (match reference) ----
#define BB   4096
#define LL   64
#define EE   384
#define HH   256
#define EPSLN 1e-5f

typedef __attribute__((ext_vector_type(16))) _Float16 v16h;
typedef __attribute__((ext_vector_type(8)))  _Float16 v8h;
typedef __attribute__((ext_vector_type(8)))  float    v8f;

// ===========================================================================
// Weight pre-pack: W [K x 256] f32 (row-major) -> f16 fragments.
// Chunk (kb*2 + half)*256 + n holds the 16 f16 values lane (half,n) feeds to
// v_wmma_f32_16x16x32_f16 for K-block kb:
//   elems 0..7  -> K = kb*32 + half*8 + e
//   elems 8..15 -> K = kb*32 + half*8 + 8 + e
// ===========================================================================
__global__ __launch_bounds__(256) void gmt_pack_w(
    const float* __restrict__ W, _Float16* __restrict__ out)
{
  const int idx  = blockIdx.x * 256 + threadIdx.x;     // (kb*2+half)*256 + n
  const int kb   = idx >> 9;
  const int half = (idx >> 8) & 1;
  const int n    = idx & 255;
  const int kbase = kb * 32 + half * 8;
  _Float16* o = out + ((size_t)idx << 4);
#pragma unroll
  for (int e = 0; e < 8; ++e)  o[e] = (_Float16)W[(size_t)(kbase + e) * HH + n];
#pragma unroll
  for (int e = 8; e < 16; ++e) o[e] = (_Float16)W[(size_t)(kbase + 8 + e) * HH + n];
}

// ===========================================================================
// Fused GEMM strip kernel: 32 rows x 256 cols per workgroup (256 thr = 8 waves)
// wave w owns N-tiles {w*16, 128+w*16} x M-tiles {0,16} -> 4 WMMAs per K-step.
// Optional: gather rows via ids, add second operand, add bias, LayerNorm(2x).
// A strip staged in LDS as f16 fragments; staging is fully vectorized:
// 8 consecutive K per task = 2x global b128 -> 4x v_cvt_pk -> 1x ds b128.
// ===========================================================================
template <int K, bool GATHER, bool HASX2, bool HASBIAS, bool DO_LN>
__global__ __launch_bounds__(256) void gmt_gemm_ln(
    const float* __restrict__ X, const float* __restrict__ X2,
    const int* __restrict__ ids,
    const _Float16* __restrict__ Wh, const float* __restrict__ bias,
    const float* __restrict__ gamma, const float* __restrict__ beta,
    float* __restrict__ Y)
{
  __shared__ __align__(32) _Float16 xh[32 * K];   // A strip, fragment layout
  __shared__ float ytile[32 * HH];
  __shared__ float red[2 * 256];
  __shared__ float mrs[2 * 32];

  const int t  = threadIdx.x;
  const int m0 = blockIdx.x * 32;

  // ---- stage + convert A strip, 8 consecutive K per task ----
  {
    constexpr int PER_M = K / 8;                 // tasks per row
    constexpr int TASKS = 32 * PER_M;
    for (int idx = t; idx < TASKS; idx += 256) {
      const int m    = idx / PER_M;
      const int rem  = idx - m * PER_M;
      const int kb   = rem >> 2;
      const int half = (rem >> 1) & 1;
      const int run  = rem & 1;
      const int k0   = kb * 32 + half * 8 + run * 16;
      const int srow = m0 + m;
      const float* src = GATHER ? (X + (size_t)ids[srow] * K)
                                : (X + (size_t)srow * K);
      float4 a = *(const float4*)(src + k0);
      float4 b = *(const float4*)(src + k0 + 4);
      if (HASX2) {
        const float* s2 = X2 + (size_t)srow * K;
        float4 a2 = *(const float4*)(s2 + k0);
        float4 b2 = *(const float4*)(s2 + k0 + 4);
        a.x += a2.x; a.y += a2.y; a.z += a2.z; a.w += a2.w;
        b.x += b2.x; b.y += b2.y; b.z += b2.z; b.w += b2.w;
      }
      v8h h;
      h[0] = (_Float16)a.x; h[1] = (_Float16)a.y;
      h[2] = (_Float16)a.z; h[3] = (_Float16)a.w;
      h[4] = (_Float16)b.x; h[5] = (_Float16)b.y;
      h[6] = (_Float16)b.z; h[7] = (_Float16)b.w;
      *(v8h*)(xh + ((((kb << 1) + half) << 5) + m) * 16 + run * 8) = h;
    }
  }
  __syncthreads();

  const int wave = t >> 5;
  const int lane = t & 31;
  const int hl   = lane >> 4;      // K-half selector for this lane
  const int ll   = lane & 15;      // M (for A) / N (for B) within tile
  const int n0a  = wave * 16;
  const int n0b  = 128 + wave * 16;

  const v16h* __restrict__ xp = (const v16h*)xh;
  const v16h* __restrict__ wp = (const v16h*)Wh;

  v8f c00 = {}, c01 = {}, c10 = {}, c11 = {};
#pragma unroll
  for (int kb = 0; kb < K / 32; ++kb) {
    const int fa = ((kb << 1) + hl) << 5;          // A frag base (32 rows)
    const int fb = ((kb << 1) + hl) << 8;          // B frag base (256 cols)
    v16h a0 = xp[fa + ll];                          // M-tile 0 (rows 0..15)
    v16h a1 = xp[fa + 16 + ll];                     // M-tile 1 (rows 16..31)
    v16h b0 = wp[fb + n0a + ll];
    v16h b1 = wp[fb + n0b + ll];
    c00 = __builtin_amdgcn_wmma_f32_16x16x32_f16(false, a0, false, b0, (short)0, c00, false, false);
    c01 = __builtin_amdgcn_wmma_f32_16x16x32_f16(false, a0, false, b1, (short)0, c01, false, false);
    c10 = __builtin_amdgcn_wmma_f32_16x16x32_f16(false, a1, false, b0, (short)0, c10, false, false);
    c11 = __builtin_amdgcn_wmma_f32_16x16x32_f16(false, a1, false, b1, (short)0, c11, false, false);
  }

  // ---- epilogue into LDS (D layout: M = r + 8*(lane>=16), N = lane&15) ----
  {
    const int n    = ll;
    const int moff = hl << 3;
    const float ba = HASBIAS ? bias[n0a + n] : 0.f;
    const float bb = HASBIAS ? bias[n0b + n] : 0.f;
#pragma unroll
    for (int r = 0; r < 8; ++r) {
      float v00 = c00[r] + ba, v01 = c01[r] + bb;
      float v10 = c10[r] + ba, v11 = c11[r] + bb;
      if (DO_LN) { v00 *= 2.f; v01 *= 2.f; v10 *= 2.f; v11 *= 2.f; }  // LN(xp+xp)
      ytile[(moff + r) * HH + n0a + n]      = v00;
      ytile[(moff + r) * HH + n0b + n]      = v01;
      ytile[(16 + moff + r) * HH + n0a + n] = v10;
      ytile[(16 + moff + r) * HH + n0b + n] = v11;
    }
  }
  __syncthreads();

  if (DO_LN) {
    // 8 threads per row, 32 elements each
    const int row = t >> 3, part = t & 7;
    float s = 0.f, ss = 0.f;
    const float* yr = ytile + row * HH + part * 32;
#pragma unroll
    for (int i = 0; i < 32; ++i) { float v = yr[i]; s += v; ss += v * v; }
    red[row * 8 + part]       = s;
    red[256 + row * 8 + part] = ss;
    __syncthreads();
    if (part == 0) {
      float sum = 0.f, sumsq = 0.f;
#pragma unroll
      for (int i = 0; i < 8; ++i) { sum += red[row * 8 + i]; sumsq += red[256 + row * 8 + i]; }
      const float mean = sum * (1.f / HH);
      const float var  = sumsq * (1.f / HH) - mean * mean;
      mrs[row]      = mean;
      mrs[32 + row] = rsqrtf(var + EPSLN);
    }
    __syncthreads();
    const float g = gamma[t], be = beta[t];
#pragma unroll
    for (int r = 0; r < 32; ++r) {
      float v = ytile[r * HH + t];
      Y[(size_t)(m0 + r) * HH + t] = g * (v - mrs[r]) * mrs[32 + r] + be;
    }
  } else {
#pragma unroll
    for (int r = 0; r < 32; ++r) {
      Y[(size_t)(m0 + r) * HH + t] = ytile[r * HH + t];
    }
  }
}

// ===========================================================================
// scores[b, j] = sum_k relu(A_l[j,k] + C[b,k]) * w2[k] + b2
// one block per batch row; C row broadcast via LDS; A_l (64 KB) is L2-hot.
// All streaming reads are float4 (b128).
// ===========================================================================
__global__ __launch_bounds__(256) void gmt_score(
    const float* __restrict__ Cmat, const float* __restrict__ A_l,
    const float* __restrict__ w2, const float* __restrict__ b2,
    float* __restrict__ out)
{
  __shared__ __align__(16) float crow[HH];
  __shared__ float partial[256];
  const int b = blockIdx.x;
  const int t = threadIdx.x;

  if (t < HH / 4) {
    ((float4*)crow)[t] = ((const float4*)(Cmat + (size_t)b * HH))[t];
  }
  __syncthreads();

  const int j   = t & 63;
  const int seg = t >> 6;
  const float4* ar4 = (const float4*)(A_l + (size_t)j * HH + seg * 64);
  const float4* cr4 = (const float4*)(crow + seg * 64);
  const float4* w24 = (const float4*)(w2 + seg * 64);
  float s = 0.f;
#pragma unroll
  for (int q = 0; q < 16; ++q) {
    const float4 av = ar4[q];
    const float4 cv = cr4[q];
    const float4 wv = w24[q];
    s += fmaxf(av.x + cv.x, 0.f) * wv.x;
    s += fmaxf(av.y + cv.y, 0.f) * wv.y;
    s += fmaxf(av.z + cv.z, 0.f) * wv.z;
    s += fmaxf(av.w + cv.w, 0.f) * wv.w;
  }
  partial[t] = s;
  __syncthreads();
  if (t < 64) {
    float r = partial[t] + partial[t + 64] + partial[t + 128] + partial[t + 192];
    out[(size_t)b * LL + t] = r + b2[0];
  }
}

// ===========================================================================
extern "C" void kernel_launch(void* const* d_in, const int* in_sizes, int n_in,
                              void* d_out, int out_size, void* d_ws, size_t ws_size,
                              hipStream_t stream)
{
  const int*   user_ids = (const int*)  d_in[0];
  const float* qemb     = (const float*)d_in[1];
  const float* lemb     = (const float*)d_in[2];
  const float* table    = (const float*)d_in[3];
  const float* Wu0 = (const float*)d_in[4];  const float* bu0 = (const float*)d_in[5];
  const float* Wq0 = (const float*)d_in[6];  const float* bq0 = (const float*)d_in[7];
  const float* Wl0 = (const float*)d_in[8];  const float* bl0 = (const float*)d_in[9];
  const float* g0  = (const float*)d_in[10]; const float* be0 = (const float*)d_in[11];
  const float* Wu1 = (const float*)d_in[12]; const float* bu1 = (const float*)d_in[13];
  const float* Wq1 = (const float*)d_in[14]; const float* bq1 = (const float*)d_in[15];
  const float* Wl1 = (const float*)d_in[16]; const float* bl1 = (const float*)d_in[17];
  const float* g1  = (const float*)d_in[18]; const float* be1 = (const float*)d_in[19];
  const float* w1  = (const float*)d_in[20]; const float* b1  = (const float*)d_in[21];
  const float* w2  = (const float*)d_in[22]; const float* b2  = (const float*)d_in[23];
  float* out = (float*)d_out;
  float* ws  = (float*)d_ws;

  const float* w1a = w1;             // rows 0..255   (llm part of combined)
  const float* w1b = w1 + 256 * HH;  // rows 256..511 (uq part)

  // ---- workspace layout (float units; all offsets 32B-aligned) ----
  const size_t BUF  = (size_t)BB * HH;              // 1,048,576 floats
  const size_t LBUF = (size_t)LL * HH;              // 16,384 floats
  float* u0   = ws + 0 * BUF;
  float* q0   = ws + 1 * BUF;
  float* u1   = ws + 2 * BUF;
  float* q1   = ws + 3 * BUF;
  float* Cmat = ws + 0 * BUF;                       // reuse u0 (dead by then)
  float* l0   = ws + 4 * BUF;
  float* l1   = ws + 4 * BUF + LBUF;
  float* A_l  = ws + 4 * BUF + 2 * LBUF;

  // packed f16 weights (counted in floats: f16 count / 2)
  const size_t PW384 = (size_t)EE * HH / 2;         // 49,152 floats
  const size_t PW256 = (size_t)HH * HH / 2;         // 32,768 floats
  float* pbase = ws + 4 * BUF + 3 * LBUF;
  _Float16* pWu0 = (_Float16*)(pbase);
  _Float16* pWq0 = (_Float16*)(pbase + 1 * PW384);
  _Float16* pWl0 = (_Float16*)(pbase + 2 * PW384);
  _Float16* pWu1 = (_Float16*)(pbase + 3 * PW384);
  _Float16* pWq1 = (_Float16*)(pbase + 3 * PW384 + 1 * PW256);
  _Float16* pWl1 = (_Float16*)(pbase + 3 * PW384 + 2 * PW256);
  _Float16* pW1a = (_Float16*)(pbase + 3 * PW384 + 3 * PW256);
  _Float16* pW1b = (_Float16*)(pbase + 3 * PW384 + 4 * PW256);

  const dim3 blk(256);
  const int PG384 = (EE / 32) * 2 * HH / 256;       // 24 blocks
  const int PG256 = (HH / 32) * 2 * HH / 256;       // 16 blocks

  // ---- pack weights to WMMA-native f16 fragments (~1.2 MB total) ----
  gmt_pack_w<<<PG384, blk, 0, stream>>>(Wu0, pWu0);
  gmt_pack_w<<<PG384, blk, 0, stream>>>(Wq0, pWq0);
  gmt_pack_w<<<PG384, blk, 0, stream>>>(Wl0, pWl0);
  gmt_pack_w<<<PG256, blk, 0, stream>>>(Wu1, pWu1);
  gmt_pack_w<<<PG256, blk, 0, stream>>>(Wq1, pWq1);
  gmt_pack_w<<<PG256, blk, 0, stream>>>(Wl1, pWl1);
  gmt_pack_w<<<PG256, blk, 0, stream>>>(w1a, pW1a);
  gmt_pack_w<<<PG256, blk, 0, stream>>>(w1b, pW1b);

  // ---- GNN layer 0 (E=384 -> H=256), LN(2x) fused ----
  gmt_gemm_ln<EE, true,  false, true, true><<<BB / 32, blk, 0, stream>>>(
      table, nullptr, user_ids, pWu0, bu0, g0, be0, u0);
  gmt_gemm_ln<EE, false, false, true, true><<<BB / 32, blk, 0, stream>>>(
      qemb, nullptr, nullptr, pWq0, bq0, g0, be0, q0);
  gmt_gemm_ln<EE, false, false, true, true><<<LL / 32, blk, 0, stream>>>(
      lemb, nullptr, nullptr, pWl0, bl0, g0, be0, l0);

  // ---- GNN layer 1 (H -> H), LN(2x) fused ----
  gmt_gemm_ln<HH, false, false, true, true><<<BB / 32, blk, 0, stream>>>(
      u0, nullptr, nullptr, pWu1, bu1, g1, be1, u1);
  gmt_gemm_ln<HH, false, false, true, true><<<BB / 32, blk, 0, stream>>>(
      q0, nullptr, nullptr, pWq1, bq1, g1, be1, q1);
  gmt_gemm_ln<HH, false, false, true, true><<<LL / 32, blk, 0, stream>>>(
      l0, nullptr, nullptr, pWl1, bl1, g1, be1, l1);

  // ---- factorized MLP: C = (u1+q1) @ w1b + b1 ; A_l = l1 @ w1a ----
  gmt_gemm_ln<HH, false, true,  true,  false><<<BB / 32, blk, 0, stream>>>(
      u1, q1, nullptr, pW1b, b1, nullptr, nullptr, Cmat);
  gmt_gemm_ln<HH, false, false, false, false><<<LL / 32, blk, 0, stream>>>(
      l1, nullptr, nullptr, pW1a, nullptr, nullptr, nullptr, A_l);

  // ---- fused relu-dot scoring ----
  gmt_score<<<BB, blk, 0, stream>>>(Cmat, A_l, w2, b2, out);
}
